// SNGNN_Plus_Plus_62689342652827
// MI455X (gfx1250) — compile-verified, matrix operand order checked by reading, not run on hardware
//
#include <hip/hip_runtime.h>
#include <stdint.h>
#include <math.h>

#define IN_DIM  256
#define HID_DIM 128
#define OUT_DIM 40
#define TOPK    2

typedef __attribute__((ext_vector_type(16))) __bf16          v16bf;
typedef __attribute__((ext_vector_type(16))) unsigned short  v16us;
typedef __attribute__((ext_vector_type(8)))  float           v8f;
typedef __attribute__((ext_vector_type(4)))  float           f4;

__device__ __forceinline__ unsigned short f32_to_bf16_rne(float f) {
    unsigned u = __float_as_uint(f);
    u += 0x7FFFu + ((u >> 16) & 1u);   // round-to-nearest-even
    return (unsigned short)(u >> 16);
}

// Monotone float -> uint32 key (order-preserving for all finite floats).
__device__ __forceinline__ unsigned enc_f32(float f) {
    unsigned u = __float_as_uint(f);
    return (u & 0x80000000u) ? ~u : (u | 0x80000000u);
}

// ---------------------------------------------------------------------------
// GEMM: Y[M x Fout] = X[M x FIN] @ W[Fout x FIN]^T + bias, bf16 WMMA, f32 acc.
// 128-thread blocks = 4 waves; each wave owns one 16x16 output tile.
// Row indices are CLAMPED (not masked): out-of-range rows/cols only pollute
// D entries we never store, so all fragment loads are unconditional b128.
// ---------------------------------------------------------------------------
template <int FIN>
__global__ __launch_bounds__(128)
void wmma_gemm_bias(const float* __restrict__ X, const float* __restrict__ W,
                    const float* __restrict__ bias, float* __restrict__ Y,
                    int M, int Fout, int tiles_n)
{
    const int lane = threadIdx.x & 31;
    const int wv   = threadIdx.x >> 5;
    const int tn   = blockIdx.y * 4 + wv;
    if (tn >= tiles_n) return;               // wave-uniform exit

    const int m0   = blockIdx.x * 16;
    const int n0   = tn * 16;
    const int half = lane >> 4;              // 0: lanes 0-15, 1: lanes 16-31
    const int sub  = lane & 15;

    const int arow   = m0 + sub;
    const int brow   = n0 + sub;
    const int arow_c = arow < M    ? arow : M - 1;
    const int brow_c = brow < Fout ? brow : Fout - 1;

    // ISA 16-bit A/B fragment layout: element j <-> K = half*8 + j (+16 for j>=8)
    const float* ap = X + (size_t)arow_c * FIN + half * 8;
    const float* bp = W + (size_t)brow_c * FIN + half * 8;

    v8f acc = {};
    #pragma unroll
    for (int k = 0; k < FIN; k += 32) {
        f4 a0 = *(const f4*)(ap + k);
        f4 a1 = *(const f4*)(ap + k + 4);
        f4 a2 = *(const f4*)(ap + k + 16);
        f4 a3 = *(const f4*)(ap + k + 20);
        f4 b0 = *(const f4*)(bp + k);
        f4 b1 = *(const f4*)(bp + k + 4);
        f4 b2 = *(const f4*)(bp + k + 16);
        f4 b3 = *(const f4*)(bp + k + 20);
        v16us ua, ub;
        #pragma unroll
        for (int j = 0; j < 4; ++j) {
            ua[j]      = f32_to_bf16_rne(a0[j]);
            ua[j + 4]  = f32_to_bf16_rne(a1[j]);
            ua[j + 8]  = f32_to_bf16_rne(a2[j]);
            ua[j + 12] = f32_to_bf16_rne(a3[j]);
            ub[j]      = f32_to_bf16_rne(b0[j]);
            ub[j + 4]  = f32_to_bf16_rne(b1[j]);
            ub[j + 8]  = f32_to_bf16_rne(b2[j]);
            ub[j + 12] = f32_to_bf16_rne(b3[j]);
        }
        acc = __builtin_amdgcn_wmma_f32_16x16x32_bf16(
                  false, __builtin_bit_cast(v16bf, ua),
                  false, __builtin_bit_cast(v16bf, ub),
                  (short)0, acc, false, false);
    }

    const int n = n0 + sub;
    if (n < Fout) {
        const float bv = bias[n];
        #pragma unroll
        for (int v = 0; v < 8; ++v) {        // D: lane = col, VGPR v -> row v + half*8
            const int m = m0 + v + half * 8;
            if (m < M) Y[(size_t)m * Fout + n] = acc[v] + bv;
        }
    }
}

// ---------------------------------------------------------------------------
// rn[n] = 1 / max(||H[n,:]||, 1e-12)          (f4 loads, C % 4 == 0)
// ---------------------------------------------------------------------------
__global__ void node_rnorm(const float* __restrict__ H, float* __restrict__ rn,
                           int Nn, int C)
{
    int n = blockIdx.x * blockDim.x + threadIdx.x;
    if (n >= Nn) return;
    const f4* p = (const f4*)(H + (size_t)n * C);
    const int G = C >> 2;
    float ss = 0.f;
    for (int g = 0; g < G; ++g) {
        f4 v = p[g];
        ss += v.x * v.x + v.y * v.y + v.z * v.z + v.w * v.w;
    }
    rn[n] = 1.0f / fmaxf(sqrtf(ss), 1e-12f);
}

// Transpose W_w (C x Nn) -> Wt (Nn x C) so per-edge gathers are contiguous.
__global__ void transpose_w(const float* __restrict__ Wsrc, float* __restrict__ Wt,
                            int Nn, int C)
{
    long long idx = (long long)blockIdx.x * blockDim.x + threadIdx.x;
    if (idx >= (long long)Nn * C) return;
    int c = (int)(idx % C);
    int n = (int)(idx / C);
    Wt[idx] = Wsrc[(size_t)c * Nn + n];
}

// ---------------------------------------------------------------------------
// out0[row[e], :] += Wt[col[e], :]  and  deg[col[e]] += 1   (wave per edge)
// ---------------------------------------------------------------------------
__global__ __launch_bounds__(256)
void edge_out0_deg(const int* __restrict__ row, const int* __restrict__ col,
                   const float* __restrict__ Wt, float* __restrict__ out0,
                   float* __restrict__ deg, int E_, int C)
{
    int wid  = (int)((blockIdx.x * (long long)blockDim.x + threadIdx.x) >> 5);
    int lane = threadIdx.x & 31;
    if (wid >= E_) return;
    int r = row[wid], c_ = col[wid];
    const f4* src = (const f4*)(Wt + (size_t)c_ * C);
    float* dst = out0 + (size_t)r * C;
    const int G = C >> 2;
    for (int g = lane; g < G; g += 32) {
        f4 v = src[g];
        int c = g << 2;
        atomicAdd(&dst[c + 0], v.x);
        atomicAdd(&dst[c + 1], v.y);
        atomicAdd(&dst[c + 2], v.z);
        atomicAdd(&dst[c + 3], v.w);
    }
    if (lane == 0) atomicAdd(&deg[c_], 1.0f);
}

// sim[e] = (H[row]·H[col]) * rn[row] * rn[col]   (wave per edge, butterfly)
__global__ __launch_bounds__(256)
void edge_sim(const int* __restrict__ row, const int* __restrict__ col,
              const float* __restrict__ H, const float* __restrict__ rn,
              float* __restrict__ sim, unsigned* __restrict__ tmpk, int E_, int C)
{
    int wid  = (int)((blockIdx.x * (long long)blockDim.x + threadIdx.x) >> 5);
    int lane = threadIdx.x & 31;
    if (wid >= E_) return;
    int r = row[wid], c_ = col[wid];
    const f4* pr = (const f4*)(H + (size_t)r  * C);
    const f4* pc = (const f4*)(H + (size_t)c_ * C);
    const int G = C >> 2;
    float acc = 0.f;
    for (int g = lane; g < G; g += 32) {
        f4 a = pr[g], b = pc[g];
        acc += a.x * b.x + a.y * b.y + a.z * b.z + a.w * b.w;
    }
    for (int off = 16; off > 0; off >>= 1) acc += __shfl_xor(acc, off, 32);
    if (lane == 0) {
        float s = acc * rn[r] * rn[c_];
        sim[wid]  = s;
        tmpk[wid] = enc_f32(s);
    }
}

__global__ void init_seg(unsigned* __restrict__ segmax, int* __restrict__ arg,
                         int Nn, int E_)
{
    int n = blockIdx.x * blockDim.x + threadIdx.x;
    if (n < Nn) { segmax[n] = 0u; arg[n] = E_; }
}

__global__ void edge_segmax(const int* __restrict__ col, const unsigned* __restrict__ tmpk,
                            unsigned* __restrict__ segmax, int E_)
{
    int e = blockIdx.x * blockDim.x + threadIdx.x;
    if (e >= E_) return;
    atomicMax(&segmax[col[e]], tmpk[e]);
}

__global__ void edge_argmin(const int* __restrict__ col, const unsigned* __restrict__ tmpk,
                            const unsigned* __restrict__ segmax, int* __restrict__ arg, int E_)
{
    int e = blockIdx.x * blockDim.x + threadIdx.x;
    if (e >= E_) return;
    int c_ = col[e];
    if (tmpk[e] == segmax[c_]) atomicMin(&arg[c_], e);
}

// Pick winner per segment, record weight, knock it out (-1.1), reset for next round.
__global__ void node_select(unsigned* __restrict__ segmax, int* __restrict__ arg,
                            const float* __restrict__ sim, unsigned* __restrict__ tmpk,
                            float* __restrict__ weight, int Nn, int E_, unsigned thr_key)
{
    int n = blockIdx.x * blockDim.x + threadIdx.x;
    if (n >= Nn) return;
    int a = arg[n];
    if (a < E_ && segmax[n] >= thr_key) {
        weight[a] = sim[a];
        tmpk[a]   = enc_f32(-1.1f);
    }
    segmax[n] = 0u;
    arg[n]    = E_;
}

// s[col[e], :] += weight[e] * H[row[e], :]   (wave per edge; skip zero weights)
__global__ __launch_bounds__(256)
void edge_msg(const int* __restrict__ row, const int* __restrict__ col,
              const float* __restrict__ H, const float* __restrict__ weight,
              float* __restrict__ sacc, int E_, int C)
{
    int wid  = (int)((blockIdx.x * (long long)blockDim.x + threadIdx.x) >> 5);
    int lane = threadIdx.x & 31;
    if (wid >= E_) return;
    float w = weight[wid];
    if (w == 0.0f) return;                    // wave-uniform
    int r = row[wid], c_ = col[wid];
    const f4* src = (const f4*)(H + (size_t)r * C);
    float* dst = sacc + (size_t)c_ * C;
    const int G = C >> 2;
    for (int g = lane; g < G; g += 32) {
        f4 v = src[g];
        int c = g << 2;
        atomicAdd(&dst[c + 0], w * v.x);
        atomicAdd(&dst[c + 1], w * v.y);
        atomicAdd(&dst[c + 2], w * v.z);
        atomicAdd(&dst[c + 3], w * v.w);
    }
}

// dst = beta*(out0 + b_w) + (1-beta)*(s / max(deg,1)), optional ReLU
__global__ void node_combine(const float* __restrict__ out0, const float* __restrict__ bw,
                             const float* __restrict__ sacc, const float* __restrict__ deg,
                             const float* __restrict__ beta, float* __restrict__ dst,
                             int Nn, int C, int do_relu)
{
    long long idx = (long long)blockIdx.x * blockDim.x + threadIdx.x;
    if (idx >= (long long)Nn * C) return;
    int c = (int)(idx % C);
    int n = (int)(idx / C);
    float b  = beta[0];
    float v0 = out0[idx] + bw[c];
    float v1 = sacc[idx] / fmaxf(deg[n], 1.0f);
    float v  = b * v0 + (1.0f - b) * v1;
    if (do_relu) v = fmaxf(v, 0.0f);
    dst[idx] = v;
}

__global__ void node_logsoftmax(const float* __restrict__ H, float* __restrict__ out,
                                int Nn, int C)
{
    int n = blockIdx.x * blockDim.x + threadIdx.x;
    if (n >= Nn) return;
    const float* p = H + (size_t)n * C;
    float mx = -INFINITY;
    for (int c = 0; c < C; ++c) mx = fmaxf(mx, p[c]);
    float se = 0.f;
    for (int c = 0; c < C; ++c) se += expf(p[c] - mx);
    float lse = logf(se);
    float* o = out + (size_t)n * C;
    for (int c = 0; c < C; ++c) o[c] = p[c] - mx - lse;
}

// ---------------------------------------------------------------------------
extern "C" void kernel_launch(void* const* d_in, const int* in_sizes, int n_in,
                              void* d_out, int out_size, void* d_ws, size_t ws_size,
                              hipStream_t stream)
{
    const int Nn = in_sizes[0] / IN_DIM;
    const int E_ = in_sizes[1] / 2;

    const float* x     = (const float*)d_in[0];
    const int*   ei    = (const int*)  d_in[1];
    const int*   row   = ei;
    const int*   col   = ei + E_;
    const float* Wlin0 = (const float*)d_in[2];
    const float* blin0 = (const float*)d_in[3];
    const float* Ww0   = (const float*)d_in[4];
    const float* bw0   = (const float*)d_in[5];
    const float* beta0 = (const float*)d_in[6];
    const float* Wlin1 = (const float*)d_in[7];
    const float* blin1 = (const float*)d_in[8];
    const float* Ww1   = (const float*)d_in[9];
    const float* bw1   = (const float*)d_in[10];
    const float* bet1  = (const float*)d_in[11];

    // Workspace carving (256B-aligned)
    char* wp = (char*)d_ws;
    auto alloc = [&](size_t bytes) -> void* {
        void* p = wp; wp += (bytes + 255) & ~(size_t)255; return p;
    };
    float*    hin    = (float*)   alloc((size_t)Nn * HID_DIM * 4); // layer input / final feats
    float*    hfeat  = (float*)   alloc((size_t)Nn * HID_DIM * 4); // h of current layer
    float*    out0   = (float*)   alloc((size_t)Nn * HID_DIM * 4);
    float*    sacc   = (float*)   alloc((size_t)Nn * HID_DIM * 4);
    float*    Wt     = (float*)   alloc((size_t)Nn * HID_DIM * 4); // transposed W_w
    float*    rn     = (float*)   alloc((size_t)Nn * 4);
    float*    deg    = (float*)   alloc((size_t)Nn * 4);
    unsigned* segmax = (unsigned*)alloc((size_t)Nn * 4);
    int*      arg    = (int*)     alloc((size_t)Nn * 4);
    float*    sim    = (float*)   alloc((size_t)E_ * 4);
    unsigned* tmpk   = (unsigned*)alloc((size_t)E_ * 4);
    float*    weight = (float*)   alloc((size_t)E_ * 4);
    (void)ws_size; (void)n_in; (void)out_size;

    const unsigned thr_key = 0x80000000u; // enc_f32(0.0f) == THR

    auto run_edges = [&](int C, const float* Ww, const float* bw,
                         const float* beta, float* dst, int do_relu)
    {
        // reciprocal row norms
        node_rnorm<<<(Nn + 255) / 256, 256, 0, stream>>>(hfeat, rn, Nn, C);
        // transpose W_w for contiguous per-edge gathers
        long long totW = (long long)Nn * C;
        transpose_w<<<(unsigned)((totW + 255) / 256), 256, 0, stream>>>(Ww, Wt, Nn, C);
        // zero accumulators
        hipMemsetAsync(out0,   0, (size_t)Nn * C * 4, stream);
        hipMemsetAsync(sacc,   0, (size_t)Nn * C * 4, stream);
        hipMemsetAsync(deg,    0, (size_t)Nn * 4,     stream);
        hipMemsetAsync(weight, 0, (size_t)E_ * 4,     stream);
        // out0 scatter + degree
        edge_out0_deg<<<(E_ + 7) / 8, 256, 0, stream>>>(row, col, Wt, out0, deg, E_, C);
        // cosine similarity per edge
        edge_sim<<<(E_ + 7) / 8, 256, 0, stream>>>(row, col, hfeat, rn, sim, tmpk, E_, C);
        // top-K selection (serialized rounds of segmax -> argmin -> select)
        init_seg<<<(Nn + 255) / 256, 256, 0, stream>>>(segmax, arg, Nn, E_);
        for (int t = 0; t < TOPK; ++t) {
            edge_segmax<<<(E_ + 255) / 256, 256, 0, stream>>>(col, tmpk, segmax, E_);
            edge_argmin<<<(E_ + 255) / 256, 256, 0, stream>>>(col, tmpk, segmax, arg, E_);
            node_select<<<(Nn + 255) / 256, 256, 0, stream>>>(segmax, arg, sim, tmpk,
                                                              weight, Nn, E_, thr_key);
        }
        // weighted messages
        edge_msg<<<(E_ + 7) / 8, 256, 0, stream>>>(row, col, hfeat, weight, sacc, E_, C);
        // blend
        long long totC = (long long)Nn * C;
        node_combine<<<(unsigned)((totC + 255) / 256), 256, 0, stream>>>(
            out0, bw, sacc, deg, beta, dst, Nn, C, do_relu);
    };

    // ---- Layer 0: 256 -> 128, ReLU ----
    {
        const int tiles_n = (HID_DIM + 15) / 16;                 // 8
        dim3 gg((Nn + 15) / 16, (tiles_n + 3) / 4);
        wmma_gemm_bias<IN_DIM><<<gg, 128, 0, stream>>>(x, Wlin0, blin0, hfeat,
                                                       Nn, HID_DIM, tiles_n);
        run_edges(HID_DIM, Ww0, bw0, beta0, hin, 1);
    }
    // ---- Layer 1: 128 -> 40 ----
    {
        const int tiles_n = (OUT_DIM + 15) / 16;                 // 3
        dim3 gg((Nn + 15) / 16, (tiles_n + 3) / 4);
        wmma_gemm_bias<HID_DIM><<<gg, 128, 0, stream>>>(hin, Wlin1, blin1, hfeat,
                                                        Nn, OUT_DIM, tiles_n);
        run_edges(OUT_DIM, Ww1, bw1, bet1, hin, 0);
    }
    // log_softmax -> output
    node_logsoftmax<<<(Nn + 255) / 256, 256, 0, stream>>>(hin, (float*)d_out, Nn, OUT_DIM);
}
